// CNN_LSTM_BiDAF_Model_76544907149888
// MI455X (gfx1250) — matrix operand order repaired and checked by reading, not compiled, not run on hardware
//
#include <hip/hip_runtime.h>

// ---------------------------------------------------------------------------
// CDNA5 (gfx1250, wave32) CNN_LSTM_BiDAF forward.
// All dense math on v_wmma_f32_16x16x32_f16; operands padded so every
// fragment load is unconditional, aligned and wide (b128).
// ---------------------------------------------------------------------------

typedef __attribute__((ext_vector_type(16))) _Float16 v16h;
typedef __attribute__((ext_vector_type(8)))  _Float16 v8h;
typedef __attribute__((ext_vector_type(8)))  float    v8f;
typedef __attribute__((ext_vector_type(4)))  float    v4f;

__device__ __forceinline__ float sigf(float x){ return 1.0f/(1.0f+__expf(-x)); }

__device__ __forceinline__ float act_apply(float v, int act){
  if(act==1) return fmaxf(v,0.0f);
  if(act==2) return sigf(v);
  if(act==3) return tanhf(v);
  return v;
}

__device__ __forceinline__ v8f wmma_f16(v16h a, v16h b, v8f c){
#if defined(__AMDGCN__)
  c = __builtin_amdgcn_wmma_f32_16x16x32_f16(false, a, false, b, (short)0, c,
                                             false, false);
#endif
  return c;
}

// A fragment (16x32 f16 per ISA 7.12.2): per lane two contiguous runs of 8
// elements at k0+hi*8 and k0+16+hi*8.  All strides are multiples of 8 floats
// and buffers 32B-aligned -> unconditional aligned vector loads.
__device__ __forceinline__ v16h load_a(const float* row, int k0, int hi){
  const v4f* p0 = (const v4f*)(row + k0 + hi*8);
  const v4f* p1 = (const v4f*)(row + k0 + 16 + hi*8);
  v4f f0 = p0[0], f1 = p0[1], g0 = p1[0], g1 = p1[1];
  v16h a;
#pragma unroll
  for(int e=0;e<4;++e){
    a[e]    = (_Float16)f0[e];
    a[4+e]  = (_Float16)f1[e];
    a[8+e]  = (_Float16)g0[e];
    a[12+e] = (_Float16)g1[e];
  }
  return a;
}

__device__ __forceinline__ v16h load_a(const _Float16* row, int k0, int hi){
  v8h lo = *(const v8h*)(row + k0 + hi*8);
  v8h hh = *(const v8h*)(row + k0 + 16 + hi*8);
  v16h a;
#pragma unroll
  for(int e=0;e<8;++e){ a[e] = lo[e]; a[8+e] = hh[e]; }
  return a;
}

// A-row address resolution:
//  mode 0: row m        -> A + m*lda
//  mode 1: conv windows -> A + (m/aG)*aGS + (m%aG)*lda
//  mode 2: time-major   -> A + (m%aG)*aGS + (m/aG)*lda
//  mode 3: gather       -> A + rowIds[m]*lda
template<typename AT>
__device__ __forceinline__ const AT* resolve_row(const AT* A, int m, int lda,
    int aMode, int aG, long long aGS, const int* rowIds){
  if(aMode==1) return A + (long long)(m / aG) * aGS + (long long)(m % aG) * lda;
  if(aMode==2) return A + (long long)(m % aG) * aGS + (long long)(m / aG) * lda;
  if(aMode==3) return A + (long long)rowIds[m] * lda;
  return A + (long long)m * lda;
}

// ---------------------------------------------------------------------------
// Generic WMMA GEMM.  M multiple of 16; Np mult of 16; Kp mult of 32.
// B is pre-packed f16 [Np][Kp] (zero padded) -> two aligned v8h loads/lane.
// ---------------------------------------------------------------------------
template<typename AT>
__global__ __launch_bounds__(256) void wmma_gemm_k(
    const AT* __restrict__ A, const _Float16* __restrict__ Bp,
    const float* __restrict__ bias, const float* __restrict__ Cadd,
    float* __restrict__ C,
    int M, int Np, int Kp,
    int lda, int aMode, int aG, long long aGS, const int* __restrict__ rowIds,
    int ldc, int Nstore,
    long long sAb, long long sBb, long long sCb,
    int act)
{
  int lane = threadIdx.x & 31;
  int wav  = threadIdx.x >> 5;
  int Nt = Np >> 4;
  int tile = blockIdx.x * 8 + wav;
  if (tile >= (M >> 4) * Nt) return;          // wave-uniform
  int tN = tile % Nt, tM = tile / Nt;
  long long batch = blockIdx.y;

  int hi  = lane >> 4;
  int l15 = lane & 15;
  int m = tM*16 + l15;
  const AT* row = resolve_row(A + batch*sAb, m, lda, aMode, aG, aGS, rowIds);
  int n = tN*16 + l15;
  const _Float16* bcol = Bp + batch*sBb + (long long)n*Kp;

  v8f acc = {};
  for(int k0 = 0; k0 < Kp; k0 += 32){
    v16h a = load_a(row, k0, hi);
    v16h b = __builtin_shufflevector(*(const v8h*)(bcol + k0 + hi*8), *(const v8h*)(bcol + k0 + 16 + hi*8), 0,1,2,3,4,5,6,7,8,9,10,11,12,13,14,15);
    acc = wmma_f16(a, b, acc);
  }

  if(n < Nstore){
    float bv = bias ? bias[n] : 0.0f;
    float* Cb = C + batch*sCb;
#pragma unroll
    for(int r=0;r<8;++r){
      int mm = tM*16 + r + hi*8;
      float v = acc[r] + bv;
      if(Cadd) v += Cadd[(long long)mm*ldc + n];
      Cb[(long long)mm*ldc + n] = act_apply(v, act);
    }
  }
}

// ---------------------------------------------------------------------------
// Fused highway layer: Y = sigmoid(X GW + GB) * relu(X LW + LB) + (1-g)*X
// Dp = 160 padded width; weights packed f16 [160][160]; fully unconditional.
// ---------------------------------------------------------------------------
__global__ __launch_bounds__(256) void highway_kernel(
    const float* __restrict__ X,
    const _Float16* __restrict__ LWp, const float* __restrict__ LBp,
    const _Float16* __restrict__ GWp, const float* __restrict__ GBp,
    float* __restrict__ Y, int M, int Dp)
{
  int lane = threadIdx.x & 31;
  int wav  = threadIdx.x >> 5;
  int Nt = Dp >> 4;
  int tile = blockIdx.x * 8 + wav;
  if (tile >= (M >> 4) * Nt) return;
  int tN = tile % Nt, tM = tile / Nt;

  int hi  = lane >> 4;
  int l15 = lane & 15;
  const float* row = X + (long long)(tM*16 + l15)*Dp;
  int n = tN*16 + l15;
  const _Float16* lcol = LWp + (long long)n*Dp;
  const _Float16* gcol = GWp + (long long)n*Dp;

  v8f ch = {}, cg = {};
  for(int k0 = 0; k0 < Dp; k0 += 32){
    v16h a  = load_a(row, k0, hi);
    v16h bl = __builtin_shufflevector(*(const v8h*)(lcol + k0 + hi*8), *(const v8h*)(lcol + k0 + 16 + hi*8), 0,1,2,3,4,5,6,7,8,9,10,11,12,13,14,15);
    v16h bg = __builtin_shufflevector(*(const v8h*)(gcol + k0 + hi*8), *(const v8h*)(gcol + k0 + 16 + hi*8), 0,1,2,3,4,5,6,7,8,9,10,11,12,13,14,15);
    ch = wmma_f16(a, bl, ch);
    cg = wmma_f16(a, bg, cg);
  }
  float lb = LBp[n], gb = GBp[n];
#pragma unroll
  for(int r=0;r<8;++r){
    int mm = tM*16 + r + hi*8;
    float h = fmaxf(ch[r] + lb, 0.0f);
    float g = sigf(cg[r] + gb);
    float x = X[(long long)mm*Dp + n];
    Y[(long long)mm*Dp + n] = g*h + (1.0f - g)*x;
  }
}

// ---------------------------------------------------------------------------
// Packing kernels (run once per call; tiny vs. the GEMM work they unlock)
// ---------------------------------------------------------------------------
// dst[n][k] = trans ? src[n*ldsrc+k] : src[k*ldsrc+n], zero padded.
__global__ void pack_b_kernel(_Float16* __restrict__ dst, const float* __restrict__ src,
                              int N, int K, int Np, int Kp, int ldsrc, int trans,
                              int batches, long long sSrc, long long sDst){
  long long i = (long long)blockIdx.x*blockDim.x + threadIdx.x;
  long long per = (long long)Np*Kp;
  if(i >= (long long)batches*per) return;
  int b = (int)(i / per);
  long long r = i % per;
  int n = (int)(r / Kp), k = (int)(r % Kp);
  float v = 0.0f;
  if(n < N && k < K)
    v = trans ? src[b*sSrc + (long long)n*ldsrc + k]
              : src[b*sSrc + (long long)k*ldsrc + n];
  dst[b*sDst + (long long)n*Kp + k] = (_Float16)v;
}

// conv weight (F, j*D) -> [Np][j*Dp] with zero gap D..Dp-1 per sub-row.
__global__ void pack_conv_kernel(_Float16* __restrict__ dst, const float* __restrict__ src,
                                 int F, int j, int D, int Dp, int Np){
  int Kp = j*Dp;
  long long i = (long long)blockIdx.x*blockDim.x + threadIdx.x;
  if(i >= (long long)Np*Kp) return;
  int n = (int)(i / Kp), k = (int)(i % Kp);
  int dj = k / Dp, d = k % Dp;
  float v = 0.0f;
  if(n < F && d < D) v = src[((long long)n*j + dj)*D + d];
  dst[i] = (_Float16)v;
}

// f32 table (R,C) -> f16 (R,Cp) zero padded (A-operand for gather GEMMs)
__global__ void pack_table_kernel(_Float16* __restrict__ dst, const float* __restrict__ src,
                                  long long R, int Cc, int Cp){
  long long i = (long long)blockIdx.x*blockDim.x + threadIdx.x;
  if(i >= R*Cp) return;
  long long r = i / Cp; int c = (int)(i % Cp);
  dst[i] = (_Float16)(c < Cc ? src[r*Cc + c] : 0.0f);
}

__global__ void pack_vec_kernel(float* __restrict__ dst, const float* __restrict__ src,
                                int N, int Np){
  int i = blockIdx.x*blockDim.x + threadIdx.x;
  if(i >= Np) return;
  dst[i] = (i < N) ? src[i] : 0.0f;
}

// ---------------------------------------------------------------------------
// Small support kernels
// ---------------------------------------------------------------------------
__global__ void fill_kernel(float* __restrict__ p, long long n, float v){
  long long i = (long long)blockIdx.x*blockDim.x + threadIdx.x;
  if(i < n) p[i] = v;
}

__global__ void gather_kernel(float* __restrict__ dst, int ldd, int coloff,
                              const float* __restrict__ src, int lds,
                              const int* __restrict__ ids, int rows, int cols){
  int i = blockIdx.x*blockDim.x + threadIdx.x;
  if(i >= rows*cols) return;
  int r = i / cols, c = i % cols;
  dst[(long long)r*ldd + coloff + c] = src[(long long)ids[r]*lds + c];
}

__global__ void pool_kernel(float* __restrict__ dst, int ldd, int coloff,
                            const float* __restrict__ src, int lds, int BU, int L, int F){
  int i = blockIdx.x*blockDim.x + threadIdx.x;
  if(i >= BU*F) return;
  int bu = i / F, f = i % F;
  const float* p = src + (long long)bu*L*lds + f;
  float mx = -1e30f;
  for(int t=0;t<L;++t) mx = fmaxf(mx, p[(long long)t*lds]);
  dst[(long long)bu*ldd + coloff + f] = mx;
}

// z layout: (nB, 4H) in a zld-stride buffer = [i | f | g | o]
__global__ void lstm_cell_kernel(const float* __restrict__ z, int zld,
                                 float* __restrict__ h, float* __restrict__ c, int hld,
                                 float* __restrict__ out,
                                 int t, int T, int dirOff, int nB, int Hh, int outLd){
  int i = blockIdx.x*blockDim.x + threadIdx.x;
  if(i >= nB*Hh) return;
  int b = i / Hh, j = i % Hh;
  const float* zr = z + (long long)b*zld;
  float iv = sigf(zr[j]);
  float fv = sigf(zr[Hh + j]);
  float gv = tanhf(zr[2*Hh + j]);
  float ov = sigf(zr[3*Hh + j]);
  int hidx = b*hld + j;
  float cv = fv * c[hidx] + iv * gv;
  float hv = ov * tanhf(cv);
  c[hidx] = cv; h[hidx] = hv;
  out[((long long)b*T + t)*outLd + dirOff + j] = hv;
}

__global__ void rowdot_kernel(float* __restrict__ out, const float* __restrict__ X,
                              int ld, const float* __restrict__ wv,
                              const float* __restrict__ bias, int rows, int D){
  int i = blockIdx.x*blockDim.x + threadIdx.x;
  if(i >= rows) return;
  const float* x = X + (long long)i*ld;
  float s = 0.0f;
  for(int d=0; d<D; ++d) s += x[d]*wv[d];
  out[i] = s + bias[0];
}

// dst = src * wv[col] for col<D, 0 in pad columns
__global__ void scale_cols_kernel(float* __restrict__ dst, const float* __restrict__ src,
                                  const float* __restrict__ wv, long long rows, int ld, int D){
  long long i = (long long)blockIdx.x*blockDim.x + threadIdx.x;
  if(i >= rows*ld) return;
  int d = (int)(i % ld);
  dst[i] = (d < D) ? src[i]*wv[d] : 0.0f;
}

__global__ void att_softmax_kernel(float* __restrict__ s,
                                   const float* __restrict__ sc, const float* __restrict__ sq,
                                   const float* __restrict__ bcq,
                                   float* __restrict__ smax, int Bb, int U, int TQ){
  int i = blockIdx.x*blockDim.x + threadIdx.x;
  if(i >= Bb*U) return;
  int b = i / U;
  float* row = s + (long long)i*TQ;
  float base = bcq[0] + sc[i];
  float mx = -1e30f;
  for(int j=0;j<TQ;++j){ float v = row[j] + base + sq[b*TQ + j]; row[j] = v; mx = fmaxf(mx, v); }
  smax[i] = mx;
  float sum = 0.0f;
  for(int j=0;j<TQ;++j){ float e = __expf(row[j]-mx); row[j] = e; sum += e; }
  float inv = 1.0f/sum;
  for(int j=0;j<TQ;++j) row[j] *= inv;
}

__global__ void batt_softmax_kernel(const float* __restrict__ smax,
                                    float* __restrict__ batt, int Bb, int U){
  int b = blockIdx.x*blockDim.x + threadIdx.x;
  if(b >= Bb) return;
  const float* r = smax + (long long)b*U;
  float mx = -1e30f;
  for(int i=0;i<U;++i) mx = fmaxf(mx, r[i]);
  float sum = 0.0f;
  for(int i=0;i<U;++i) sum += __expf(r[i]-mx);
  float inv = 1.0f/sum;
  for(int i=0;i<U;++i) batt[(long long)b*U+i] = __expf(r[i]-mx)*inv;
}

__global__ void q2c_kernel(float* __restrict__ q2c, const float* __restrict__ batt,
                           const float* __restrict__ bid, int ldb_, int Bb, int U, int D){
  int i = blockIdx.x*blockDim.x + threadIdx.x;
  if(i >= Bb*D) return;
  int b = i / D, d = i % D;
  float s = 0.0f;
  for(int u=0;u<U;++u) s += batt[(long long)b*U+u]*bid[((long long)b*U+u)*ldb_ + d];
  q2c[i] = s;
}

__global__ void build_G_kernel(float* __restrict__ G, int gld,
                               const float* __restrict__ bid, int bld,
                               const float* __restrict__ c2q, int cld,
                               const float* __restrict__ q2c,
                               int Bb, int U, int D){
  int i = blockIdx.x*blockDim.x + threadIdx.x;
  if(i >= Bb*U*D) return;
  int r = i / D, d = i % D;
  int b = r / U;
  float bd = bid[(long long)r*bld + d];
  float cq = c2q[(long long)r*cld + d];
  float q  = q2c[(long long)b*D + d];
  float* g = G + (long long)r*gld;
  g[d]       = bd;
  g[D + d]   = cq;
  g[2*D + d] = bd*cq;
  g[3*D + d] = bd*q;
}

__global__ void p_kernel(float* __restrict__ p, int pld, int coloff,
                         const float* __restrict__ G, int gld,
                         const float* __restrict__ gw, const float* __restrict__ gb,
                         const float* __restrict__ Mm, int mld,
                         const float* __restrict__ mw, const float* __restrict__ mb,
                         int rows, int Dg, int Dm, int U){
  int i = blockIdx.x*blockDim.x + threadIdx.x;
  if(i >= rows) return;
  const float* gr = G + (long long)i*gld;
  const float* mr = Mm + (long long)i*mld;
  float s = gb[0] + mb[0];
  for(int d=0; d<Dg; ++d) s += gr[d]*gw[d];
  for(int d=0; d<Dm; ++d) s += mr[d]*mw[d];
  p[(long long)(i/U)*pld + coloff + (i%U)] = s;
}

__global__ void final_kernel(const float* __restrict__ logits, const float* __restrict__ mask,
                             float* __restrict__ out, int Bb, int NCc){
  int b = blockIdx.x*blockDim.x + threadIdx.x;
  if(b >= Bb) return;
  const float* lr = logits + (long long)b*NCc;
  const float* mr = mask  + (long long)b*NCc;
  float mx = -1e30f;
  for(int c=0;c<NCc;++c) mx = fmaxf(mx, lr[c]);
  float e[64];
  float sum = 0.0f;
  for(int c=0;c<NCc;++c){ e[c] = __expf(lr[c]-mx); sum += e[c]; }
  float inv = 1.0f/sum;
  float msum = 0.0f;
  for(int c=0;c<NCc;++c){ e[c] = e[c]*inv*mr[c]; msum += e[c]; }
  float im = 1.0f/msum;
  for(int c=0;c<NCc;++c){
    float v = e[c]*im;
    v = fminf(fmaxf(v, 1e-7f), 1.0f - 1e-7f);
    out[(long long)b*NCc + c] = v;
  }
}

// ---------------------------------------------------------------------------
// Host orchestration
// ---------------------------------------------------------------------------
extern "C" void kernel_launch(void* const* d_in, const int* in_sizes, int n_in,
                              void* d_out, int out_size, void* d_ws, size_t ws_size,
                              hipStream_t stream) {
  (void)in_sizes; (void)n_in; (void)out_size; (void)ws_size;
  const int Bb=64, U=64, TU=48, TQ=32, E=100, F=50, H=75, NCc=64, D2=150;
  const int Dp=160;          // D2 padded
  const int Gp=304;          // 4H padded
  const int Hp=96;           // H padded
  const int Ep=128;          // E padded
  const int GDp=608;         // 4*D2 padded
  const int Fp=64;           // F padded

  const int*   utter_ids  = (const int*)d_in[0];
  const int*   query_ids  = (const int*)d_in[1];
  const float* class_mask = (const float*)d_in[2];
  const float* emb_word   = (const float*)d_in[3];
  const float* emb_char_u = (const float*)d_in[4];
  const float* emb_char_q = (const float*)d_in[5];
  const float* conv_q_w   = (const float*)d_in[6];
  const float* conv_q_b   = (const float*)d_in[7];
  const float* conv_u_w   = (const float*)d_in[8];
  const float* conv_u_b   = (const float*)d_in[9];
  const float *hw_lw[2], *hw_lb[2], *hw_gw[2], *hw_gb[2];
  for(int l=0;l<2;++l){
    hw_lw[l]=(const float*)d_in[10+4*l]; hw_lb[l]=(const float*)d_in[11+4*l];
    hw_gw[l]=(const float*)d_in[12+4*l]; hw_gb[l]=(const float*)d_in[13+4*l];
  }
  const float *cw[3], *cb[3];
  for(int j=0;j<3;++j){ cw[j]=(const float*)d_in[18+2*j]; cb[j]=(const float*)d_in[19+2*j]; }
  const float* L[5][6];   // scene, query, lstm1, lstm2, lstm_out
  for(int k=0;k<5;++k) for(int q=0;q<6;++q) L[k][q]=(const float*)d_in[24 + 6*k + q];
  const float* att_w_c  = (const float*)d_in[54];
  const float* att_b_c  = (const float*)d_in[55];
  const float* att_w_q  = (const float*)d_in[56];
  const float* att_b_q  = (const float*)d_in[57];
  const float* att_w_cq = (const float*)d_in[58];
  const float* att_b_cq = (const float*)d_in[59];
  const float* p1g_w = (const float*)d_in[60]; const float* p1g_b = (const float*)d_in[61];
  const float* p1m_w = (const float*)d_in[62]; const float* p1m_b = (const float*)d_in[63];
  const float* p2g_w = (const float*)d_in[64]; const float* p2g_b = (const float*)d_in[65];
  const float* p2m_w = (const float*)d_in[66]; const float* p2m_b = (const float*)d_in[67];
  const float* dense_w = (const float*)d_in[68];
  const float* dense_b = (const float*)d_in[69];
  float* out = (float*)d_out;

  const int uRows = Bb*U*TU;     // 196608 (mult of 16)
  const int qRows = Bb*TQ;       // 2048

  // ---- f32 workspace (16-float aligned allocs) ----------------------------
  float* wsf = (float*)d_ws;
  size_t off = 0;
  auto alloc = [&](size_t n){ n = (n + 15) & ~(size_t)15; float* p = wsf + off; off += n; return p; };
  float* ucatA = alloc((size_t)uRows*Dp);
  float* ucatB = alloc((size_t)uRows*Dp);
  float* qcatA = alloc((size_t)qRows*Dp);
  float* qcatB = alloc((size_t)qRows*Dp);
  float* convbuf = alloc((size_t)Bb*U*(TU-1)*Fp);
  float* utter = alloc((size_t)Bb*U*Dp);
  float* XwF = alloc((size_t)U*Bb*Gp);
  float* XwB = alloc((size_t)U*Bb*Gp);
  float* zF  = alloc((size_t)Bb*Gp);
  float* zB  = alloc((size_t)Bb*Gp);
  float* hF = alloc((size_t)Bb*Hp); float* cF = alloc((size_t)Bb*Hp);
  float* hB = alloc((size_t)Bb*Hp); float* cB = alloc((size_t)Bb*Hp);
  float* bid = alloc((size_t)Bb*U*Dp);
  float* biq = alloc((size_t)Bb*TQ*Dp);
  float* s_c = alloc((size_t)Bb*U);
  float* s_q = alloc((size_t)Bb*TQ);
  float* bidw = alloc((size_t)Bb*U*Dp);
  float* sbuf = alloc((size_t)Bb*U*TQ);
  float* smax = alloc((size_t)Bb*U);
  float* batt = alloc((size_t)Bb*U);
  float* c2q  = alloc((size_t)Bb*U*Dp);
  float* q2c  = alloc((size_t)Bb*D2);
  float* Gbuf = alloc((size_t)Bb*U*GDp);
  float* tmp1 = alloc((size_t)Bb*U*Dp);
  float* Mbuf = alloc((size_t)Bb*U*Dp);
  float* M2buf= alloc((size_t)Bb*U*Dp);
  float* pbuf = alloc((size_t)Bb*2*U);
  float* logits = alloc((size_t)Bb*NCc);
  // padded bias vectors
  float* cqb_p = alloc(Fp); float* cub_p = alloc(Fp);
  float* hwlb_p[2]; float* hwgb_p[2];
  for(int l=0;l<2;++l){ hwlb_p[l]=alloc(Dp); hwgb_p[l]=alloc(Dp); }
  float* cvb_p[3]; for(int j=0;j<3;++j) cvb_p[j]=alloc(Fp);
  float* lb_p[5][2]; for(int k=0;k<5;++k){ lb_p[k][0]=alloc(Gp); lb_p[k][1]=alloc(Gp); }

  // ---- f16 packed region (32B aligned allocs) ------------------------------
  _Float16* wsh = (_Float16*)(wsf + off);
  size_t offh = 0;
  auto allocH = [&](size_t n){ n = (n + 15) & ~(size_t)15; _Float16* p = wsh + offh; offh += n; return p; };
  _Float16* embU_h = allocH((size_t)32000*Ep);
  _Float16* embQ_h = allocH((size_t)32000*Ep);
  _Float16* cqw_h = allocH((size_t)Fp*Ep);
  _Float16* cuw_h = allocH((size_t)Fp*Ep);
  _Float16* hwl_h[2]; _Float16* hwg_h[2];
  for(int l=0;l<2;++l){ hwl_h[l]=allocH((size_t)Dp*Dp); hwg_h[l]=allocH((size_t)Dp*Dp); }
  _Float16* cvw_h[3];
  for(int j=0;j<3;++j) cvw_h[j]=allocH((size_t)Fp*(j+2)*Dp);
  _Float16* Wih_h[5][2]; _Float16* Whh_h[5][2];
  int inKp[5] = {Dp, Dp, GDp, Dp, Dp};
  for(int k=0;k<5;++k) for(int d=0;d<2;++d){
    Wih_h[k][d]=allocH((size_t)Gp*inKp[k]);
    Whh_h[k][d]=allocH((size_t)Gp*Hp);
  }
  _Float16* cqB_h  = allocH((size_t)Bb*TQ*Dp);   // per-batch [32][160]
  _Float16* c2qB_h = allocH((size_t)Bb*Dp*TQ);   // per-batch [160][32]
  _Float16* dense_h = allocH((size_t)NCc*2*U);

  auto launch1 = [&](long long n){ return (unsigned)((n + 255)/256); };
  auto packB = [&](_Float16* dst, const float* src, int N,int K,int Np,int Kp,int ld,int tr,
                   int batches=1, long long sS=0, long long sD=0){
    pack_b_kernel<<<launch1((long long)batches*Np*Kp),256,0,stream>>>(dst,src,N,K,Np,Kp,ld,tr,batches,sS,sD);
  };
  auto packV = [&](float* dst, const float* src, int N, int Np){
    pack_vec_kernel<<<1,((Np+63)/64)*64,0,stream>>>(dst,src,N,Np);
  };

  // ---- weight packing ------------------------------------------------------
  pack_table_kernel<<<launch1((long long)32000*Ep),256,0,stream>>>(embU_h, emb_char_u, 32000, E, Ep);
  pack_table_kernel<<<launch1((long long)32000*Ep),256,0,stream>>>(embQ_h, emb_char_q, 32000, E, Ep);
  packB(cqw_h, conv_q_w, F, E, Fp, Ep, E, 1);
  packB(cuw_h, conv_u_w, F, E, Fp, Ep, E, 1);
  packV(cqb_p, conv_q_b, F, Fp);  packV(cub_p, conv_u_b, F, Fp);
  for(int l=0;l<2;++l){
    packB(hwl_h[l], hw_lw[l], D2, D2, Dp, Dp, D2, 0);   // x @ LW : (K,N) layout
    packB(hwg_h[l], hw_gw[l], D2, D2, Dp, Dp, D2, 0);
    packV(hwlb_p[l], hw_lb[l], D2, Dp); packV(hwgb_p[l], hw_gb[l], D2, Dp);
  }
  for(int jj=0;jj<3;++jj){
    pack_conv_kernel<<<launch1((long long)Fp*(jj+2)*Dp),256,0,stream>>>(cvw_h[jj], cw[jj], F, jj+2, D2, Dp, Fp);
    packV(cvb_p[jj], cb[jj], F, Fp);
  }
  int inD[5] = {D2, D2, 4*D2, D2, D2};
  for(int k=0;k<5;++k) for(int d=0;d<2;++d){
    packB(Wih_h[k][d], L[k][3*d+0], 4*H, inD[k], Gp, inKp[k], inD[k], 1);
    packB(Whh_h[k][d], L[k][3*d+1], 4*H, H,      Gp, Hp,      H,      1);
    packV(lb_p[k][d],  L[k][3*d+2], 4*H, Gp);
  }
  packB(dense_h, dense_w, NCc, 2*U, NCc, 2*U, NCc, 0);

  // ---- GEMM launchers ------------------------------------------------------
  auto gemmF = [&](const float* A, const _Float16* Bp, const float* bias, const float* Cadd,
                   float* C, int M,int Np,int Kp,int lda,int aMode,int aG,long long aGS,
                   const int* rowIds,int ldc,int Nstore,int act,
                   int batches=1,long long sA=0,long long sB=0,long long sC=0){
    dim3 grid((unsigned)(((M>>4)*(Np>>4) + 7)/8), (unsigned)batches);
    wmma_gemm_k<float><<<grid,256,0,stream>>>(A,Bp,bias,Cadd,C,M,Np,Kp,lda,aMode,aG,aGS,
                                              rowIds,ldc,Nstore,sA,sB,sC,act);
  };
  auto gemmH = [&](const _Float16* A, const _Float16* Bp, const float* bias, float* C,
                   int M,int Np,int Kp,int lda,const int* rowIds,int ldc,int Nstore,int act){
    dim3 grid((unsigned)(((M>>4)*(Np>>4) + 7)/8), 1);
    wmma_gemm_k<_Float16><<<grid,256,0,stream>>>(A,Bp,bias,nullptr,C,M,Np,Kp,lda,3,1,0,
                                                 rowIds,ldc,Nstore,0,0,0,act);
  };
  auto zero = [&](float* p, long long n){ fill_kernel<<<launch1(n),256,0,stream>>>(p,n,0.0f); };

  // ---- char conv + word emb + highway (query & scene) ----------------------
  gather_kernel<<<launch1((long long)qRows*E),256,0,stream>>>(qcatA, Dp, 0, emb_word, E, query_ids, qRows, E);
  gemmH(embQ_h, cqw_h, cqb_p, qcatA + E, qRows, Fp, Ep, Ep, query_ids, Dp, F, 1);
  unsigned qblk = (unsigned)(((qRows>>4)*(Dp>>4) + 7)/8);
  highway_kernel<<<qblk,256,0,stream>>>(qcatA, hwl_h[0],hwlb_p[0],hwg_h[0],hwgb_p[0], qcatB, qRows, Dp);
  highway_kernel<<<qblk,256,0,stream>>>(qcatB, hwl_h[1],hwlb_p[1],hwg_h[1],hwgb_p[1], qcatA, qRows, Dp);

  gather_kernel<<<launch1((long long)uRows*E),256,0,stream>>>(ucatA, Dp, 0, emb_word, E, utter_ids, uRows, E);
  gemmH(embU_h, cuw_h, cub_p, ucatA + E, uRows, Fp, Ep, Ep, utter_ids, Dp, F, 1);
  unsigned ublk = (unsigned)(((uRows>>4)*(Dp>>4) + 7)/8);
  highway_kernel<<<ublk,256,0,stream>>>(ucatA, hwl_h[0],hwlb_p[0],hwg_h[0],hwgb_p[0], ucatB, uRows, Dp);
  highway_kernel<<<ublk,256,0,stream>>>(ucatB, hwl_h[1],hwlb_p[1],hwg_h[1],hwgb_p[1], ucatA, uRows, Dp);

  // ---- text convolutions (strided-window GEMMs over padded rows) + pool ----
  for(int jj=0;jj<3;++jj){
    int j = 2 + jj;
    int Lw = TU - j + 1;
    int M = Bb*U*Lw;
    gemmF(ucatA, cvw_h[jj], cvb_p[jj], nullptr, convbuf,
          M, Fp, j*Dp, Dp, 1, Lw, (long long)TU*Dp, nullptr, Fp, Fp, 1);
    pool_kernel<<<launch1(Bb*U*F),256,0,stream>>>(utter, Dp, jj*F, convbuf, Fp, Bb*U, Lw, F);
  }

  // ---- BiLSTM driver -------------------------------------------------------
  auto run_bilstm = [&](const float* x, int T, int li, float* outb){
    int Kin = inKp[li];
    gemmF(x, Wih_h[li][0], lb_p[li][0], nullptr, XwF, T*Bb, Gp, Kin, Kin, 2, Bb, (long long)T*Kin, nullptr, Gp, Gp, 0);
    gemmF(x, Wih_h[li][1], lb_p[li][1], nullptr, XwB, T*Bb, Gp, Kin, Kin, 2, Bb, (long long)T*Kin, nullptr, Gp, Gp, 0);
    zero(hF, Bb*Hp); zero(cF, Bb*Hp); zero(hB, Bb*Hp); zero(cB, Bb*Hp);
    for(int s=0;s<T;++s){
      int tf = s, tb = T-1-s;
      gemmF(hF, Whh_h[li][0], nullptr, XwF + (long long)tf*Bb*Gp, zF, Bb, Gp, Hp, Hp, 0,1,0,nullptr, Gp, Gp, 0);
      gemmF(hB, Whh_h[li][1], nullptr, XwB + (long long)tb*Bb*Gp, zB, Bb, Gp, Hp, Hp, 0,1,0,nullptr, Gp, Gp, 0);
      lstm_cell_kernel<<<launch1(Bb*H),256,0,stream>>>(zF,Gp,hF,cF,Hp,outb,tf,T,0,Bb,H,Dp);
      lstm_cell_kernel<<<launch1(Bb*H),256,0,stream>>>(zB,Gp,hB,cB,Hp,outb,tb,T,H,Bb,H,Dp);
    }
  };

  run_bilstm(utter, U,  0, bid);   // lstm_scene -> bi_d (B,U,160-padded)
  run_bilstm(qcatA, TQ, 1, biq);   // lstm_query -> bi_q

  // ---- BiDAF attention -----------------------------------------------------
  rowdot_kernel<<<launch1(Bb*U),256,0,stream>>>(s_c, bid, Dp, att_w_c, att_b_c, Bb*U, D2);
  rowdot_kernel<<<launch1(Bb*TQ),256,0,stream>>>(s_q, biq, Dp, att_w_q, att_b_q, Bb*TQ, D2);
  scale_cols_kernel<<<launch1((long long)Bb*U*Dp),256,0,stream>>>(bidw, bid, att_w_cq, Bb*U, Dp, D2);
  // pack bi_q two ways (activation B operands)
  packB(cqB_h,  biq, TQ, D2, TQ, Dp, Dp, 1, Bb, (long long)TQ*Dp, (long long)TQ*Dp);
  packB(c2qB_h, biq, D2, TQ, Dp, TQ, Dp, 0, Bb, (long long)TQ*Dp, (long long)Dp*TQ);
  // cq[b] = (bid*w) @ biq[b]^T
  gemmF(bidw, cqB_h, nullptr, nullptr, sbuf, U, TQ, Dp, Dp, 0,1,0,nullptr, TQ, TQ, 0,
        Bb, (long long)U*Dp, (long long)TQ*Dp, (long long)U*TQ);
  att_softmax_kernel<<<launch1(Bb*U),256,0,stream>>>(sbuf, s_c, s_q, att_b_cq, smax, Bb, U, TQ);
  batt_softmax_kernel<<<1,64,0,stream>>>(smax, batt, Bb, U);
  // c2q[b] = a @ biq[b]
  gemmF(sbuf, c2qB_h, nullptr, nullptr, c2q, U, Dp, TQ, TQ, 0,1,0,nullptr, Dp, Dp, 0,
        Bb, (long long)U*TQ, (long long)Dp*TQ, (long long)U*Dp);
  q2c_kernel<<<launch1(Bb*D2),256,0,stream>>>(q2c, batt, bid, Dp, Bb, U, D2);
  build_G_kernel<<<launch1((long long)Bb*U*D2),256,0,stream>>>(Gbuf, GDp, bid, Dp, c2q, Dp, q2c, Bb, U, D2);

  // ---- modeling layers -----------------------------------------------------
  run_bilstm(Gbuf, U, 2, tmp1);    // lstm1 (in=600)
  run_bilstm(tmp1, U, 3, Mbuf);    // lstm2
  p_kernel<<<launch1(Bb*U),256,0,stream>>>(pbuf, 2*U, 0, Gbuf, GDp, p1g_w, p1g_b,
                                           Mbuf, Dp, p1m_w, p1m_b, Bb*U, 4*D2, D2, U);
  run_bilstm(Mbuf, U, 4, M2buf);   // lstm_out
  p_kernel<<<launch1(Bb*U),256,0,stream>>>(pbuf, 2*U, U, Gbuf, GDp, p2g_w, p2g_b,
                                           M2buf, Dp, p2m_w, p2m_b, Bb*U, 4*D2, D2, U);

  // ---- classifier ----------------------------------------------------------
  gemmF(pbuf, dense_h, dense_b, nullptr, logits, Bb, NCc, 2*U, 2*U, 0,1,0,nullptr, NCc, NCc, 0);
  final_kernel<<<1,64,0,stream>>>(logits, class_mask, out, Bb, NCc);
}
